// BaseModel_46119358824625
// MI455X (gfx1250) — matrix operand order repaired and checked by reading, not compiled, not run on hardware
//
#include <hip/hip_runtime.h>

// ---------------------------------------------------------------------------
// SPINN forward for MI455X (gfx1250, wave32, WMMA + TDM).
// All GEMMs: v_wmma_f32_16x16x32_bf16 (bf16 in, f32 accumulate).
// Weights pre-packed into per-lane-contiguous WMMA B-fragment order and
// streamed into a 2x16KB LDS double buffer with tensor_load_to_lds (TDM),
// synchronized via s_wait_tensorcnt + workgroup barriers.
// ---------------------------------------------------------------------------

typedef __attribute__((ext_vector_type(16))) __bf16 v16bf;
typedef __attribute__((ext_vector_type(8)))  __bf16 v8bf;
typedef __attribute__((ext_vector_type(8)))  float  v8f;
typedef __attribute__((ext_vector_type(4)))  unsigned int v4u;
typedef __attribute__((ext_vector_type(8)))  int v8i;
typedef __attribute__((ext_vector_type(4)))  int v4i;

#define DEV __device__ __forceinline__

#if defined(__HIP_DEVICE_COMPILE__) && __has_builtin(__builtin_amdgcn_tensor_load_to_lds) && __has_builtin(__builtin_amdgcn_s_wait_tensorcnt)
#define HAVE_TDM 1
#else
#define HAVE_TDM 0
#endif

static DEV unsigned short f2bfbits(float x){
  union { float f; unsigned u; } v; v.f = x;
  unsigned r = v.u + 0x7FFFu + ((v.u >> 16) & 1u);   // round-to-nearest-even
  return (unsigned short)(r >> 16);
}

static DEV v8f vzero8(){ v8f v = {0.f,0.f,0.f,0.f,0.f,0.f,0.f,0.f}; return v; }

static DEV float sigf(float x){ return 1.0f / (1.0f + __expf(-x)); }

static DEV v8f wmma_bf16(v16bf a, v16bf b, v8f c){
  // 8 args: (neg_a, A, neg_b, B, c_mod, C, reuse_a, reuse_b)
  return __builtin_amdgcn_wmma_f32_16x16x32_bf16(false, a, false, b, (short)0, c,
                                                 false, false);
}

// A fragment (16x32 bf16, ISA 7.12.2): lane holds row = lane%16, K-half kh=lane/16,
// elements 0..7 = K in [k0+8*kh, +8), elements 8..15 = K in [k0+16+8*kh, +8).
static DEV v16bf load_a_frag(const unsigned short* rowPtr, int k0, int kh){
  v16bf r;
  __builtin_memcpy(&r, rowPtr + k0 + 8*kh, 16);
  __builtin_memcpy(reinterpret_cast<char*>(&r) + 16, rowPtr + k0 + 16 + 8*kh, 16);
  return r;
}

// Packed B fragment from global: 32 contiguous bytes per lane per (ntile, kt).
static DEV v16bf load_b_frag(const unsigned short* __restrict__ W,
                             int ntile, int kt, int nKt, int lane){
  v16bf r;
  __builtin_memcpy(&r, W + ((ntile*nKt + kt)*32 + lane)*16, 32);
  return r;
}

// Packed B fragment from LDS (after TDM staged it): 32 contiguous bytes/lane.
static DEV v16bf load_b_lds(const unsigned short* p){
  v16bf r; __builtin_memcpy(&r, p, 32); return r;
}

static DEV unsigned lds_off(const void* p){
  // generic 'shared' pointer: low 32 bits carry the LDS byte address
  return (unsigned)(uintptr_t)p;
}

#if HAVE_TDM
// s_wait_tensorcnt needs a literal immediate -> macro, not a function arg.
#define tdm_wait_le(n) __builtin_amdgcn_s_wait_tensorcnt(n)

// Issue one TDM transfer: 16 rows of 1KB (one fragment-tile each), row stride
// strideElems bf16 elements, gathered into contiguous LDS at ldsOff.
// D# per CDNA5 ISA ch.7/8: group0 = {flags, lds_addr, global_addr, type},
// group1 = {data_size/tensor dims/tile dims/strides}, groups 2-3 zero (2-D).
// This toolchain's builtin is the 6-arg form:
//   (uint32x4 g0, int32x8 g1, int32x4 g2, int32x4 g3, int32x8 g4, i32 cpol)
static DEV void tdm_issue(const unsigned short* src, unsigned ldsOff,
                          unsigned strideElems){
  unsigned long long ga = (unsigned long long)(uintptr_t)src;
  unsigned su = strideElems >> 2;                 // stride in 8-byte units
  v4u g0;
  g0[0] = 1u;                                     // count=1, valid user D#
  g0[1] = ldsOff;                                 // lds_addr (bytes)
  g0[2] = (unsigned)ga;                           // global_addr[31:0]
  g0[3] = (unsigned)((ga >> 32) & 0x01FFFFFFu)    // global_addr[56:32]
        | (2u << 30);                             // type = 2 ("image")
  v8i g1;
  g1[0] = 3 << 16;                                // data_size = 3 (8B); mask=0
  g1[1] = (int)(0xFFFFu << 16);                   // tensor_dim0 lo16 (huge)
  g1[2] = (int)(0x000Fu | (0xFFFFu << 16));       // dim0 hi16 | dim1 lo16
  g1[3] = (int)(0x000Fu | (128u << 16));          // dim1 hi16 | tile_dim0=128
  g1[4] = 16;                                     // tile_dim1=16, tile_dim2=0
  g1[5] = (int)su;                                // tensor_dim0_stride lo32
  g1[6] = 0;                                      // stride hi16 | dim1_stride lo
  g1[7] = 0;
  v4i gz4 = {0,0,0,0};
  v8i gz8 = {0,0,0,0,0,0,0,0};
  __builtin_amdgcn_tensor_load_to_lds(g0, g1, gz4, gz4, gz8, 0);
}
#endif

// Fallback: cooperative synchronous copy of one 16-tile chunk into LDS.
static DEV void copy_chunk(unsigned short* dstLds, const unsigned short* src,
                           unsigned strideElems, int tid){
  for (int i = tid; i < 16*512; i += 1024){
    int row = i >> 9, e = i & 511;
    dstLds[i] = src[row*strideElems + e];
  }
}

// ---------------------------------------------------------------------------
__global__ void zero_f32_kernel(float* __restrict__ p, long n){
  for (long i = blockIdx.x*(long)blockDim.x + threadIdx.x; i < n;
       i += (long)gridDim.x*blockDim.x) p[i] = 0.f;
}

// Pack row-major (K x N) f32 weight into bf16 WMMA B-fragment order.
__global__ void pack_w_kernel(const float* __restrict__ src,
                              unsigned short* __restrict__ dst,
                              int Ksrc, int Kpad, int N){
  int nKt = Kpad >> 5;
  int total = Kpad * N;
  for (int i = blockIdx.x*blockDim.x + threadIdx.x; i < total;
       i += gridDim.x*blockDim.x){
    int e    = i & 15;
    int lane = (i >> 4) & 31;
    int r2   = i >> 9;
    int kt   = r2 % nKt;
    int nt   = r2 / nKt;
    int K    = kt*32 + ((e < 8) ? (8*(lane >> 4) + e) : (16 + 8*(lane >> 4) + (e - 8)));
    int col  = nt*16 + (lane & 15);
    float v  = (K < Ksrc) ? src[K*N + col] : 0.f;
    dst[i] = f2bfbits(v);
  }
}

// ---------------------------------------------------------------------------
// bufs = embed_table[tokens] @ W_proj : (16384 x 512), K = 300 (padded 320).
__global__ __launch_bounds__(256) void embed_proj_kernel(
    const int* __restrict__ tokens, const float* __restrict__ embed,
    const unsigned short* __restrict__ Wp, float* __restrict__ bufs){
  int lane = threadIdx.x & 31;
  int gw   = blockIdx.x * 8 + (threadIdx.x >> 5);
  int tm   = gw >> 5;
  int tn   = gw & 31;
  int kh   = lane >> 4;
  int arow = tm*16 + (lane & 15);
  int tok  = tokens[arow];
  const float* erow = embed + (long)tok * 300;
  v8f acc = vzero8();
  for (int kt = 0; kt < 10; ++kt){
    int k0 = kt*32;
    unsigned short bits[16];
    #pragma unroll
    for (int e = 0; e < 16; ++e){
      int K = (e < 8) ? (k0 + 8*kh + e) : (k0 + 16 + 8*kh + (e - 8));
      float v = (K < 300) ? erow[K] : 0.f;
      bits[e] = f2bfbits(v);
    }
    v16bf a; __builtin_memcpy(&a, bits, 32);
    acc = wmma_bf16(a, load_b_frag(Wp, tn, kt, 10, lane), acc);
  }
  int colb = tn*16 + (lane & 15);
  #pragma unroll
  for (int r = 0; r < 8; ++r){
    int m = tm*16 + 8*kh + r;
    bufs[(long)m*512 + colb] = acc[r];
  }
}

// ---------------------------------------------------------------------------
// Persistent single-workgroup sequential scan (255 steps).
// A staging (global bf16): 128 x 832 = [buf_h | s1_h | s2_h | h].
// Weight fragments streamed via TDM into LDS double buffer.
__global__ __launch_bounds__(1024) void spinn_scan_kernel(
    const int* __restrict__ trans, const float* __restrict__ bufs,
    float* __restrict__ stack, unsigned short* __restrict__ Astage,
    float* __restrict__ c_glob,
    const unsigned short* __restrict__ Wb_p,  const unsigned short* __restrict__ Ws1_p,
    const unsigned short* __restrict__ Ws2_p, const unsigned short* __restrict__ Wl_p,
    const unsigned short* __restrict__ Wleft_p, const unsigned short* __restrict__ Wright_p,
    const unsigned short* __restrict__ Wtrack_p,
    const float* __restrict__ bl, const float* __restrict__ b_red,
    int* __restrict__ spf){
  __shared__ unsigned short wtile[2][16*512];     // 2 x 16KB TDM weight buffers
  __shared__ int sp_s[128], bp_s[128], tr_s[128];
  const int tid  = threadIdx.x;
  const int lane = tid & 31;
  const int w    = tid >> 5;
  const int kh   = lane >> 4;

  for (int i = tid; i < 128*64; i += 1024){
    c_glob[i] = 0.f;
    Astage[(i >> 6)*832 + 768 + (i & 63)] = 0;    // h0 = 0 (bf16)
  }
  if (tid < 128){ sp_s[tid] = 0; bp_s[tid] = 0; }
  __threadfence();
  __syncthreads();

  for (int t = 0; t < 255; ++t){
    if (tid < 128) tr_s[tid] = trans[tid*255 + t];
    __syncthreads();

#if HAVE_TDM
    // kick off first phase-1 weight chunk; DMA overlaps A assembly
    if (w == 0) tdm_issue(Wb_p, lds_off(&wtile[0][0]), 8*512);
#endif

    // ---- assemble A: buf_h | s1_h | s2_h (h region persists) ----
    for (int i = tid; i < 128*768; i += 1024){
      int b = i / 768, k = i - b*768;
      int sp = sp_s[b];
      float v;
      if (k < 256){
        int bp = bp_s[b]; if (bp > 127) bp = 127;
        v = bufs[(b*128 + bp)*512 + k];
      } else if (k < 512){
        int kk = k - 256;
        v = (sp >= 1) ? stack[(b*129 + (sp - 1))*512 + kk] : 0.f;
      } else {
        int kk = k - 512;
        v = (sp >= 2) ? stack[(b*129 + (sp - 2))*512 + kk] : 0.f;
      }
      Astage[b*832 + k] = f2bfbits(v);
    }
    __threadfence();
    __syncthreads();

    // ---- phase 1: gates(128x256) = A x [Wb;Ws1;Ws2;Wl], then LSTM cell ----
    {
      const unsigned short* Wsrc[4] = { Wb_p, Ws1_p, Ws2_p, Wl_p };
      const int aOff1[4] = { 0, 256, 512, 768 };
      const int nkts1[4] = { 8, 8, 8, 2 };
      const int cb1[4]   = { 0, 8, 16, 24 };
      int mb = w >> 2, jb = w & 3;                // 8 row-blocks x 4 col-quads
      const unsigned short* Arow = Astage + (mb*16 + (lane & 15))*832;
      v8f acc[4];
      #pragma unroll
      for (int g = 0; g < 4; ++g) acc[g] = vzero8();

      for (int c = 0; c < 26; ++c){
        int s  = (c < 8) ? 0 : (c < 16) ? 1 : (c < 24) ? 2 : 3;
        int kt = c - cb1[s];
#if HAVE_TDM
        if (w == 0){
          int cn = c + 1;
          if (cn < 26){
            int sn  = (cn < 8) ? 0 : (cn < 16) ? 1 : (cn < 24) ? 2 : 3;
            int ktn = cn - cb1[sn];
            tdm_issue(Wsrc[sn] + ktn*512, lds_off(&wtile[cn & 1][0]),
                      (unsigned)nkts1[sn]*512);
            tdm_wait_le(1);                       // chunk c complete
          } else {
            tdm_wait_le(0);
          }
        }
        __syncthreads();
        const unsigned short* bbuf = &wtile[c & 1][0];
#else
        copy_chunk(&wtile[0][0], Wsrc[s] + kt*512, (unsigned)nkts1[s]*512, tid);
        __syncthreads();
        const unsigned short* bbuf = &wtile[0][0];
#endif
        v16bf a = load_a_frag(Arow, aOff1[s] + kt*32, kh);
        #pragma unroll
        for (int g = 0; g < 4; ++g)               // a,i,f,o tiles share A frag
          acc[g] = wmma_bf16(a, load_b_lds(bbuf + (g*4 + jb)*512 + lane*16), acc[g]);
        __syncthreads();                          // buffer reuse fence
      }

#if HAVE_TDM
      // prefetch first phase-2 chunk while the cell epilogue runs
      if (w == 0) tdm_issue(Wleft_p, lds_off(&wtile[0][0]), 8*512);
#endif
      int j = jb*16 + (lane & 15);                // 0..63 within gate block
      #pragma unroll
      for (int r = 0; r < 8; ++r){
        int m = mb*16 + 8*kh + r;
        float av = acc[0][r] + bl[j];
        float iv = acc[1][r] + bl[64 + j];
        float fv = acc[2][r] + bl[128 + j];
        float ov = acc[3][r] + bl[192 + j];
        float cp = c_glob[m*64 + j];
        float cn = tanhf(av)*sigf(iv) + sigf(fv)*cp;
        float hn = sigf(ov)*tanhf(cn);
        c_glob[m*64 + j] = cn;
        Astage[m*832 + 768 + j] = f2bfbits(hn);   // new h for phase 2 / next t
      }
    }
    __threadfence();
    __syncthreads();

    // ---- phase 2: r_in(128x1280) = s2h@Wleft + s1h@Wright + h@Wtrack ----
    {
      const unsigned short* Wsrc[3] = { Wleft_p, Wright_p, Wtrack_p };
      const int aOff2[3] = { 512, 256, 768 };
      const int nkts2[3] = { 8, 8, 2 };
      v8f acc2[4][5];
      #pragma unroll
      for (int i4 = 0; i4 < 4; ++i4)
        #pragma unroll
        for (int g = 0; g < 5; ++g) acc2[i4][g] = vzero8();

      for (int c = 0; c < 90; ++c){               // (seg, kt, gate) chunks
        int s = (c < 40) ? 0 : (c < 80) ? 1 : 2;
        int l = c - ((s == 0) ? 0 : (s == 1) ? 40 : 80);
        int kt = l / 5, g = l - kt*5;
#if HAVE_TDM
        if (w == 0){
          int cn = c + 1;
          if (cn < 90){
            int sn = (cn < 40) ? 0 : (cn < 80) ? 1 : 2;
            int ln = cn - ((sn == 0) ? 0 : (sn == 1) ? 40 : 80);
            int ktn = ln / 5, gn = ln - ktn*5;
            tdm_issue(Wsrc[sn] + (gn*16*nkts2[sn] + ktn)*512,
                      lds_off(&wtile[cn & 1][0]), (unsigned)nkts2[sn]*512);
            tdm_wait_le(1);
          } else {
            tdm_wait_le(0);
          }
        }
        __syncthreads();
        const unsigned short* bbuf = &wtile[c & 1][0];
#else
        copy_chunk(&wtile[0][0], Wsrc[s] + (g*16*nkts2[s] + kt)*512,
                   (unsigned)nkts2[s]*512, tid);
        __syncthreads();
        const unsigned short* bbuf = &wtile[0][0];
#endif
        #pragma unroll
        for (int i4 = 0; i4 < 4; ++i4){
          int task = i4*32 + w;                   // 128 tasks = 8 mb x 16 nt
          int mb2 = task >> 4, nt = task & 15;
          const unsigned short* Arow2 = Astage + (mb2*16 + (lane & 15))*832;
          v16bf a  = load_a_frag(Arow2, aOff2[s] + kt*32, kh);
          v16bf bf = load_b_lds(bbuf + nt*512 + lane*16);
          #pragma unroll
          for (int gg = 0; gg < 5; ++gg)          // keep acc index compile-time
            if (gg == g) acc2[i4][gg] = wmma_bf16(a, bf, acc2[i4][gg]);
        }
        __syncthreads();                          // buffer reuse fence
      }

      // TreeLSTM reduce epilogue
      #pragma unroll
      for (int i4 = 0; i4 < 4; ++i4){
        int task = i4*32 + w;
        int mb2 = task >> 4, nt = task & 15;
        int j = nt*16 + (lane & 15);              // 0..255 within H
        #pragma unroll
        for (int r = 0; r < 8; ++r){
          int m  = mb2*16 + 8*kh + r;             // batch row
          int sp = sp_s[m];
          float av  = acc2[i4][0][r] + b_red[j];
          float iv  = acc2[i4][1][r] + b_red[256 + j];
          float flv = acc2[i4][2][r] + b_red[512 + j];
          float frv = acc2[i4][3][r] + b_red[768 + j];
          float ov  = acc2[i4][4][r] + b_red[1024 + j];
          float c1 = (sp >= 1) ? stack[(m*129 + (sp - 1))*512 + 256 + j] : 0.f;
          float c2 = (sp >= 2) ? stack[(m*129 + (sp - 2))*512 + 256 + j] : 0.f;
          float cr = tanhf(av)*sigf(iv) + sigf(flv)*c2 + sigf(frv)*c1;
          float hr = sigf(ov)*tanhf(cr);
          if (tr_s[m] != 0){                      // REDUCE: write [h_red|c_red]
            int wp = sp - 2; if (wp < 0) wp = 0;
            float* dst = stack + (m*129 + wp)*512;
            dst[j]       = hr;
            dst[256 + j] = cr;
          }
        }
      }
    }

    // ---- SHIFT rows: push buf_top (full 2H, f32) ----
    for (int i = tid; i < 128*512; i += 1024){
      int b = i >> 9, k = i & 511;
      if (tr_s[b] == 0){
        int bp = bp_s[b]; if (bp > 127) bp = 127;
        stack[(b*129 + sp_s[b])*512 + k] = bufs[(b*128 + bp)*512 + k];
      }
    }
    __threadfence();
    __syncthreads();
    if (tid < 128){
      int sh = (tr_s[tid] == 0);
      sp_s[tid] += sh ? 1 : -1;
      bp_s[tid] += sh;
    }
    __syncthreads();
  }

  if (tid < 128) spf[tid] = sp_s[tid];
}

// ---------------------------------------------------------------------------
// hid = relu(top[:, :H] @ W1 + b1)   (128 x 1024, K = 256)
__global__ __launch_bounds__(1024) void classifier1_kernel(
    const float* __restrict__ stack, const int* __restrict__ spf,
    const unsigned short* __restrict__ W1p, const float* __restrict__ b1,
    unsigned short* __restrict__ Fstage, float* __restrict__ hid){
  const int tid = threadIdx.x, lane = tid & 31, w = tid >> 5, kh = lane >> 4;
  for (int i = tid; i < 128*256; i += 1024){
    int b = i >> 8, k = i & 255;
    int sp = spf[b] - 1; if (sp < 0) sp = 0;
    Fstage[i] = f2bfbits(stack[(b*129 + sp)*512 + k]);
  }
  __threadfence();
  __syncthreads();
  for (int it = 0; it < 16; ++it){
    int task = it*32 + w;                          // 512 tasks = 8 mb x 64 nt
    int mb = task >> 6, nt = task & 63;
    const unsigned short* Arow = Fstage + (mb*16 + (lane & 15))*256;
    v8f acc = vzero8();
    for (int kt = 0; kt < 8; ++kt){
      v16bf a = load_a_frag(Arow, kt*32, kh);
      acc = wmma_bf16(a, load_b_frag(W1p, nt, kt, 8, lane), acc);
    }
    int col = nt*16 + (lane & 15);
    #pragma unroll
    for (int r = 0; r < 8; ++r){
      int m = mb*16 + 8*kh + r;
      float v = acc[r] + b1[col];
      hid[m*1024 + col] = v > 0.f ? v : 0.f;
    }
  }
}

// out = hid @ W2 + b2   (128 x 3) — tiny, scalar f32.
__global__ void classifier2_kernel(const float* __restrict__ hid,
    const float* __restrict__ W2, const float* __restrict__ b2,
    float* __restrict__ out){
  int i = blockIdx.x*blockDim.x + threadIdx.x;
  if (i >= 128*3) return;
  int b = i / 3, c = i - b*3;
  float s = b2[c];
  const float* hrow = hid + b*1024;
  for (int m = 0; m < 1024; ++m) s += hrow[m] * W2[m*3 + c];
  out[i] = s;
}

// ---------------------------------------------------------------------------
extern "C" void kernel_launch(void* const* d_in, const int* in_sizes, int n_in,
                              void* d_out, int out_size, void* d_ws, size_t ws_size,
                              hipStream_t stream){
  (void)in_sizes; (void)n_in; (void)out_size; (void)ws_size;
  const int*   tokens = (const int*)  d_in[0];
  const int*   trans  = (const int*)  d_in[1];
  const float* embed  = (const float*)d_in[2];
  const float* W_proj = (const float*)d_in[3];
  const float* Wl     = (const float*)d_in[4];
  const float* bl     = (const float*)d_in[5];
  const float* Wb     = (const float*)d_in[6];
  const float* Ws1    = (const float*)d_in[7];
  const float* Ws2    = (const float*)d_in[8];
  const float* Wleft  = (const float*)d_in[9];
  const float* Wright = (const float*)d_in[10];
  const float* Wtrack = (const float*)d_in[11];
  const float* b_red  = (const float*)d_in[12];
  const float* W1     = (const float*)d_in[13];
  const float* b1     = (const float*)d_in[14];
  const float* W2     = (const float*)d_in[15];
  const float* b2     = (const float*)d_in[16];
  float* out = (float*)d_out;

  char* ws = (char*)d_ws;
  size_t off = 0;
  auto carve = [&](size_t bytes)->char*{
    char* p = ws + off;
    off += (bytes + 255) & ~size_t(255);
    return p;
  };
  float*          bufs     = (float*)         carve(16384ull*512*4);
  float*          stack    = (float*)         carve(128ull*129*512*4);
  unsigned short* Astage   = (unsigned short*)carve(128ull*832*2);
  unsigned short* Fstage   = (unsigned short*)carve(128ull*256*2);
  float*          hid      = (float*)         carve(128ull*1024*4);
  float*          c_glob   = (float*)         carve(128ull*64*4);
  int*            spf      = (int*)           carve(128*4);
  unsigned short* Wproj_p  = (unsigned short*)carve(320ull*512*2);
  unsigned short* Wb_p     = (unsigned short*)carve(256ull*256*2);
  unsigned short* Ws1_p    = (unsigned short*)carve(256ull*256*2);
  unsigned short* Ws2_p    = (unsigned short*)carve(256ull*256*2);
  unsigned short* Wl_p     = (unsigned short*)carve(64ull*256*2);
  unsigned short* Wleft_p  = (unsigned short*)carve(256ull*1280*2);
  unsigned short* Wright_p = (unsigned short*)carve(256ull*1280*2);
  unsigned short* Wtrack_p = (unsigned short*)carve(64ull*1280*2);
  unsigned short* W1_p     = (unsigned short*)carve(256ull*1024*2);

  hipLaunchKernelGGL(zero_f32_kernel, dim3(2048), dim3(256), 0, stream,
                     stack, (long)128*129*512);

  hipLaunchKernelGGL(pack_w_kernel, dim3(320), dim3(256), 0, stream, W_proj, Wproj_p, 300, 320, 512);
  hipLaunchKernelGGL(pack_w_kernel, dim3(320), dim3(256), 0, stream, Wb,     Wb_p,    256, 256, 256);
  hipLaunchKernelGGL(pack_w_kernel, dim3(320), dim3(256), 0, stream, Ws1,    Ws1_p,   256, 256, 256);
  hipLaunchKernelGGL(pack_w_kernel, dim3(320), dim3(256), 0, stream, Ws2,    Ws2_p,   256, 256, 256);
  hipLaunchKernelGGL(pack_w_kernel, dim3(320), dim3(256), 0, stream, Wl,     Wl_p,     64,  64, 256);
  hipLaunchKernelGGL(pack_w_kernel, dim3(320), dim3(256), 0, stream, Wleft,  Wleft_p, 256, 256, 1280);
  hipLaunchKernelGGL(pack_w_kernel, dim3(320), dim3(256), 0, stream, Wright, Wright_p,256, 256, 1280);
  hipLaunchKernelGGL(pack_w_kernel, dim3(320), dim3(256), 0, stream, Wtrack, Wtrack_p, 64,  64, 1280);
  hipLaunchKernelGGL(pack_w_kernel, dim3(320), dim3(256), 0, stream, W1,     W1_p,    256, 256, 1024);

  hipLaunchKernelGGL(embed_proj_kernel, dim3(4096), dim3(256), 0, stream,
                     tokens, embed, Wproj_p, bufs);

  hipLaunchKernelGGL(spinn_scan_kernel, dim3(1), dim3(1024), 0, stream,
                     trans, bufs, stack, Astage, c_glob,
                     Wb_p, Ws1_p, Ws2_p, Wl_p, Wleft_p, Wright_p, Wtrack_p,
                     bl, b_red, spf);

  hipLaunchKernelGGL(classifier1_kernel, dim3(1), dim3(1024), 0, stream,
                     stack, spf, W1_p, b1, Fstage, hid);
  hipLaunchKernelGGL(classifier2_kernel, dim3(2), dim3(192), 0, stream,
                     hid, W2, b2, out);
}
